// LocalPooledPointNet2d_40544491274267
// MI455X (gfx1250) — matrix operand order repaired and checked by reading, not compiled, not run on hardware
//
#include <hip/hip_runtime.h>

// ---------------- problem constants ----------------
#define B_    4
#define N_    32768
#define BN    131072        // B_*N_  (2^17)
#define DIM   128
#define RES   64
#define R2    4096
#define NBLK  5

typedef __attribute__((ext_vector_type(16))) __bf16 v16bf;
typedef __attribute__((ext_vector_type(8)))  __bf16 v8bf;
typedef __attribute__((ext_vector_type(8)))  float  v8f;

// ---------------- bf16 helpers (storage-only, no __bf16 arithmetic) -------
__device__ __forceinline__ __bf16 f2bf(float f) {
    union { float f; unsigned u; } v; v.f = f;
    unsigned r = (v.u + 0x7fffu + ((v.u >> 16) & 1u)) >> 16;   // RNE
    union { unsigned short s; __bf16 b; } o; o.s = (unsigned short)r;
    return o.b;
}
__device__ __forceinline__ float bf2f(__bf16 b) {
    union { __bf16 b; unsigned short s; } i; i.b = b;
    union { unsigned u; float f; } o; o.u = ((unsigned)i.s) << 16;
    return o.f;
}

// order-preserving float<->uint for atomic max (u==0 is the -inf sentinel)
__device__ __forceinline__ unsigned f2ord(float f) {
    union { float f; unsigned u; } v; v.f = f;
    return (v.u & 0x80000000u) ? ~v.u : (v.u | 0x80000000u);
}
__device__ __forceinline__ float ord2f(unsigned u) {
    union { unsigned u; float f; } v;
    v.u = (u & 0x80000000u) ? (u & 0x7fffffffu) : ~u;
    return v.f;
}

// ---------------- WMMA fragment loaders ----------------
// A (16x32 bf16): lanes 0-15 hold row M=lane, K {k0..k0+7, k0+16..k0+23};
// lanes 16-31 hold row M=lane-16, K {k0+8..k0+15, k0+24..k0+31}.
__device__ __forceinline__ v16bf load_frag_a(const __bf16* base, int stride,
                                             int row0, int k0, int lane) {
    int half = lane >> 4, m = lane & 15;
    const __bf16* p = base + (row0 + m) * stride + k0 + half * 8;
    v8bf lo = *(const v8bf*)p;
    v8bf hi = *(const v8bf*)(p + 16);
    return __builtin_shufflevector(lo, hi, 0,1,2,3,4,5,6,7,8,9,10,11,12,13,14,15);
}

// B (32x16 bf16) from transposed weights WT[N][K] row-major:
// lanes 0-15 hold column N=n0+lane, K {k0..k0+7, k0+16..k0+23}; lanes 16-31 K {k0+8..k0+15, k0+24..k0+31}.
__device__ __forceinline__ v16bf load_frag_b(const __bf16* wt, int kstride,
                                             int n0, int k0, int lane) {
    int half = lane >> 4, n = lane & 15;
    const __bf16* p = wt + (n0 + n) * kstride + k0 + half * 8;
    v8bf lo = *(const v8bf*)p;
    v8bf hi = *(const v8bf*)(p + 16);
    return __builtin_shufflevector(lo, hi, 0,1,2,3,4,5,6,7,8,9,10,11,12,13,14,15);
}

__device__ __forceinline__ v16bf relu_frag(v16bf a) {
    union { v16bf v; unsigned short u[16]; } x; x.v = a;
#pragma unroll
    for (int i = 0; i < 16; ++i)
        if (x.u[i] & 0x8000u) x.u[i] = 0;      // bf16 negative <=> sign bit
    return x.v;
}

#define WMMA_BF16(a, b, c) \
    __builtin_amdgcn_wmma_f32_16x16x32_bf16(false, (a), false, (b), (short)0, (c), false, false)

// ---------------- weight prep: fp32 [nmat][K][N] -> bf16 [nmat][N][K] -----
__global__ void wprep_kernel(const float* __restrict__ w, __bf16* __restrict__ wt,
                             int nmat, int K, int Nc) {
    int t = blockIdx.x * 256 + threadIdx.x;
    int per = K * Nc;
    if (t >= nmat * per) return;
    int mat = t / per, r = t - mat * per;
    int k = r / Nc, n = r - k * Nc;
    wt[mat * per + n * K + k] = f2bf(w[t]);
}

// ---------------- per-point prep: xyz copy + tri-plane cell indices -------
__global__ void prep_points_kernel(const float* __restrict__ x,
                                   int* __restrict__ idx, float* __restrict__ xyz_out) {
    int p = blockIdx.x * 256 + threadIdx.x;           // 512 blocks, exact
    float v[3] = { x[3*p], x[3*p+1], x[3*p+2] };
    xyz_out[3*p] = v[0]; xyz_out[3*p+1] = v[1]; xyz_out[3*p+2] = v[2];
    const float inv = 1.0f / 1.101f;                  // 1/(1+PAD+0.001)
    const int pa[3] = {0, 0, 1}, pb[3] = {2, 1, 2};
#pragma unroll
    for (int pl = 0; pl < 3; ++pl) {
        float a = v[pa[pl]] * inv + 0.5f;
        float b = v[pb[pl]] * inv + 0.5f;
        a = fminf(fmaxf(a, 0.f), 0.999999f);
        b = fminf(fmaxf(b, 0.f), 0.999999f);
        int i = (int)(a * (float)RES); i = i < 0 ? 0 : (i > RES-1 ? RES-1 : i);
        int j = (int)(b * (float)RES); j = j < 0 ? 0 : (j > RES-1 ? RES-1 : j);
        idx[pl * BN + p] = i + RES * j;
    }
}

// ---------------- stem: x(3) @ W(3x256) + b -> hw bf16 (BN x 256) ---------
__global__ __launch_bounds__(256) void stem_kernel(const float* __restrict__ x,
        const float* __restrict__ w, const float* __restrict__ b,
        __bf16* __restrict__ hw) {
    __shared__ float sw[3 * 256];
    __shared__ float sb[256];
    int tid = threadIdx.x;
    for (int i = tid; i < 768; i += 256) sw[i] = w[i];
    sb[tid] = b[tid];
    __syncthreads();
    long p = (long)blockIdx.x * 8 + (tid >> 5);       // 8 points / block
    int g = tid & 31;                                 // 32 channel-groups of 8
    float x0 = x[p*3], x1 = x[p*3+1], x2 = x[p*3+2];
#pragma unroll
    for (int j = 0; j < 8; ++j) {
        int ch = g * 8 + j;
        float v = sb[ch] + x0 * sw[ch] + x1 * sw[256 + ch] + x2 * sw[512 + ch];
        hw[p * 256 + ch] = f2bf(v);
    }
}

// ---------------- fused residual block -----------------------------------
//  in : hw  bf16 (BN x 256)                        (h, or concat[h,pool])
//  out: hn  bf16 (BN x 128)
//  out = h @ Ws + bs + relu( relu(h) @ W0 + b0 ) @ W1 + b1
//
//  GEMM order chosen so the two 64-VGPR accumulators are never live at the
//  same time; B-fragments are batch-loaded (16 b128 loads in flight) so the
//  scheduler can wait incrementally instead of draining loadcnt per WMMA.
#define TM 64
__global__ __launch_bounds__(128) void block_kernel(
        const __bf16* __restrict__ hw, __bf16* __restrict__ hn,
        const __bf16* __restrict__ w0t, const float* __restrict__ b0,
        const __bf16* __restrict__ w1t, const float* __restrict__ b1,
        const __bf16* __restrict__ wst, const float* __restrict__ bs) {
    __shared__ alignas(16) __bf16 sh[TM * 256];       // 32 KB input tile
    __shared__ alignas(16) __bf16 snet[TM * 128];     // 16 KB relu(net) tile
    const int  tid  = threadIdx.x;
    const int  lane = tid & 31;
    const int  wave = tid >> 5;
    const int  half = lane >> 4;
    const int  col  = lane & 15;
    const long rowBase = (long)blockIdx.x * TM;

    { // cooperative tile load (16B per thread per iter, coalesced)
        const uint4* src = (const uint4*)(hw + rowBase * 256);
        uint4* dst = (uint4*)sh;
        for (int i = tid; i < TM * 256 / 8; i += 128) dst[i] = src[i];
    }
    __syncthreads();

    const int m0 = wave * 16;                          // this wave's rows

    // ---- net = relu(h) @ W0   (K = 256) ----
    v8f acc[8] = {};
#pragma unroll
    for (int kc = 0; kc < 256; kc += 32) {
        v16bf a = relu_frag(load_frag_a(sh, 256, m0, kc, lane));
        v16bf bf[8];
#pragma unroll
        for (int nt = 0; nt < 8; ++nt)
            bf[nt] = load_frag_b(w0t, 256, nt * 16, kc, lane);
#pragma unroll
        for (int nt = 0; nt < 8; ++nt)
            acc[nt] = WMMA_BF16(a, bf[nt], acc[nt]);
    }
    // relu(net + b0) -> snet (C-layout -> A-layout via LDS; same-wave rows only)
#pragma unroll
    for (int nt = 0; nt < 8; ++nt) {
        int n = nt * 16 + col;
        float bias = b0[n];
#pragma unroll
        for (int r = 0; r < 8; ++r) {
            float v = acc[nt][r] + bias;
            v = v > 0.f ? v : 0.f;
            snet[(m0 + r + 8 * half) * 128 + n] = f2bf(v);
        }
    }

    // ---- shortcut: acc_sc = h @ Ws   (K = 256) ----  (acc now dead)
    v8f acc_sc[8] = {};
#pragma unroll
    for (int kc = 0; kc < 256; kc += 32) {
        v16bf a = load_frag_a(sh, 256, m0, kc, lane);
        v16bf bf[8];
#pragma unroll
        for (int nt = 0; nt < 8; ++nt)
            bf[nt] = load_frag_b(wst, 256, nt * 16, kc, lane);
#pragma unroll
        for (int nt = 0; nt < 8; ++nt)
            acc_sc[nt] = WMMA_BF16(a, bf[nt], acc_sc[nt]);
    }

    // ---- dx = relu(net) @ W1, accumulated into acc_sc (K = 128) ----
#pragma unroll
    for (int kc = 0; kc < 128; kc += 32) {
        v16bf a = load_frag_a(snet, 128, m0, kc, lane);
        v16bf bf[8];
#pragma unroll
        for (int nt = 0; nt < 8; ++nt)
            bf[nt] = load_frag_b(w1t, 128, nt * 16, kc, lane);
#pragma unroll
        for (int nt = 0; nt < 8; ++nt)
            acc_sc[nt] = WMMA_BF16(a, bf[nt], acc_sc[nt]);
    }

    // ---- epilogue: out = acc_sc + bs + b1 -> hn bf16 ----
#pragma unroll
    for (int nt = 0; nt < 8; ++nt) {
        int n = nt * 16 + col;
        float bias = bs[n] + b1[n];
#pragma unroll
        for (int r = 0; r < 8; ++r) {
            long row = rowBase + m0 + r + 8 * half;
            hn[row * 128 + n] = f2bf(acc_sc[nt][r] + bias);
        }
    }
}

// ---------------- head: c = hn @ Wc + bc (fp32 out to d_out) --------------
__global__ __launch_bounds__(128) void head_kernel(
        const __bf16* __restrict__ hn, const __bf16* __restrict__ wct,
        const float* __restrict__ bc, float* __restrict__ cout) {
    __shared__ alignas(16) __bf16 sh[TM * 128];
    const int  tid  = threadIdx.x;
    const int  lane = tid & 31;
    const int  wave = tid >> 5;
    const int  half = lane >> 4;
    const int  col  = lane & 15;
    const long rowBase = (long)blockIdx.x * TM;
    {
        const uint4* src = (const uint4*)(hn + rowBase * 128);
        uint4* dst = (uint4*)sh;
        for (int i = tid; i < TM * 128 / 8; i += 128) dst[i] = src[i];
    }
    __syncthreads();
    const int m0 = wave * 16;
    v8f acc[8] = {};
#pragma unroll
    for (int kc = 0; kc < 128; kc += 32) {
        v16bf a = load_frag_a(sh, 128, m0, kc, lane);
        v16bf bf[8];
#pragma unroll
        for (int nt = 0; nt < 8; ++nt)
            bf[nt] = load_frag_b(wct, 128, nt * 16, kc, lane);
#pragma unroll
        for (int nt = 0; nt < 8; ++nt)
            acc[nt] = WMMA_BF16(a, bf[nt], acc[nt]);
    }
#pragma unroll
    for (int nt = 0; nt < 8; ++nt) {
        int n = nt * 16 + col;
        float bias = bc[n];
#pragma unroll
        for (int r = 0; r < 8; ++r) {
            long row = rowBase + m0 + r + 8 * half;
            cout[row * 128 + n] = acc[nt][r] + bias;
        }
    }
}

// ---------------- pooling: scatter-max into per-plane grids ---------------
// grid: uint32 order-encoded fp32, [3][B_][R2][128]; 0 == empty (-inf)
__global__ void scatter_max_kernel(const __bf16* __restrict__ hn,
                                   const int* __restrict__ idx,
                                   unsigned* __restrict__ grid) {
    long t = (long)blockIdx.x * 256 + threadIdx.x;    // 3*BN*32 threads
    int  g  = (int)(t & 31);
    long pp = t >> 5;
    int  plane = (int)(pp >> 17);                      // BN = 2^17
    long p  = pp & (BN - 1);
    int  b  = (int)(p >> 15);                          // N_ = 2^15
    int  cell = idx[plane * BN + p];
    unsigned* dst = grid + ((long)((plane * B_ + b) * R2 + cell)) * 128 + g * 4;
    const __bf16* src = hn + p * 128 + g * 4;
#pragma unroll
    for (int j = 0; j < 4; ++j)
        atomicMax(dst + j, f2ord(bf2f(src[j])));
}

// gather pooled maxima (sum over 3 planes) and build concat tile in hw
__global__ void gather_concat_kernel(const unsigned* __restrict__ grid,
                                     const int* __restrict__ idx,
                                     const __bf16* __restrict__ hn,
                                     __bf16* __restrict__ hw) {
    long t = (long)blockIdx.x * 256 + threadIdx.x;    // BN*32 threads
    int  g = (int)(t & 31);
    long p = t >> 5;
    int  b = (int)(p >> 15);
    float acc[4] = {0.f, 0.f, 0.f, 0.f};
#pragma unroll
    for (int plane = 0; plane < 3; ++plane) {
        int cell = idx[plane * BN + p];
        const unsigned* src = grid + ((long)((plane * B_ + b) * R2 + cell)) * 128 + g * 4;
#pragma unroll
        for (int j = 0; j < 4; ++j) {
            unsigned u = src[j];
            if (u != 0u) acc[j] += ord2f(u);           // empty -> contributes 0
        }
    }
#pragma unroll
    for (int j = 0; j < 4; ++j) {
        int ch = g * 4 + j;
        hw[p * 256 + 128 + ch] = f2bf(acc[j]);         // pooled half
        hw[p * 256 + ch]       = hn[p * 128 + ch];     // copy h half
    }
}

// ---------------- tri-plane mean scatter ----------------------------------
// tri layout (3,B,128,64,64): offset = ((plane*B + b)*128 + ch)*4096 + cell
__global__ void tri_scatter_kernel(const float* __restrict__ c,
                                   const int* __restrict__ idx,
                                   float* __restrict__ tri,
                                   float* __restrict__ cnt) {
    long t = (long)blockIdx.x * 256 + threadIdx.x;    // 3*BN*32 threads
    int  g  = (int)(t & 31);
    long pp = t >> 5;
    int  plane = (int)(pp >> 17);
    long p  = pp & (BN - 1);
    int  b  = (int)(p >> 15);
    int  cell = idx[plane * BN + p];
    int  pb = plane * B_ + b;
#pragma unroll
    for (int j = 0; j < 4; ++j) {
        int ch = g * 4 + j;
        atomicAdd(tri + ((long)(pb * 128 + ch)) * R2 + cell, c[p * 128 + ch]);
    }
    if (g == 0) atomicAdd(cnt + pb * R2 + cell, 1.0f);
}

__global__ void tri_final_kernel(float* __restrict__ tri, const float* __restrict__ cnt) {
    int e = blockIdx.x * 256 + threadIdx.x;           // 6291456 elems, exact
    int pb   = e >> 19;                               // / (128*4096)
    int cell = e & (R2 - 1);
    tri[e] /= fmaxf(cnt[pb * R2 + cell], 1.0f);
}

// ---------------- host-side orchestration ---------------------------------
extern "C" void kernel_launch(void* const* d_in, const int* in_sizes, int n_in,
                              void* d_out, int out_size, void* d_ws, size_t ws_size,
                              hipStream_t stream) {
    (void)in_sizes; (void)n_in; (void)out_size; (void)ws_size;
    const float* x      = (const float*)d_in[0];
    const float* stem_w = (const float*)d_in[1];
    const float* stem_b = (const float*)d_in[2];
    const float* fc0_w  = (const float*)d_in[3];
    const float* fc0_b  = (const float*)d_in[4];
    const float* fc1_w  = (const float*)d_in[5];
    const float* fc1_b  = (const float*)d_in[6];
    const float* sc_w   = (const float*)d_in[7];
    const float* sc_b   = (const float*)d_in[8];
    const float* fcc_w  = (const float*)d_in[9];
    const float* fcc_b  = (const float*)d_in[10];

    // workspace layout (bytes)
    char* ws = (char*)d_ws;
    const size_t OFF_HW   = 0;                                   // 64 MB bf16 (BN x 256)
    const size_t OFF_HN   = OFF_HW  + (size_t)BN * 256 * 2;      // 32 MB bf16 (BN x 128)
    const size_t OFF_W0T  = OFF_HN  + (size_t)BN * 128 * 2;      // 5 x 128 x 256 bf16
    const size_t OFF_WST  = OFF_W0T + (size_t)5 * 256 * 128 * 2;
    const size_t OFF_W1T  = OFF_WST + (size_t)5 * 256 * 128 * 2; // 5 x 128 x 128 bf16
    const size_t OFF_WCT  = OFF_W1T + (size_t)5 * 128 * 128 * 2; // 128 x 128 bf16
    const size_t OFF_IDX  = OFF_WCT + (size_t)128 * 128 * 2;     // 3 x BN int32
    const size_t OFF_GRID = OFF_IDX + (size_t)3 * BN * 4;        // 3 x B x R2 x 128 u32
    const size_t OFF_CNT  = OFF_GRID + (size_t)3 * B_ * R2 * 128 * 4;

    __bf16*   hw   = (__bf16*)(ws + OFF_HW);
    __bf16*   hn   = (__bf16*)(ws + OFF_HN);
    __bf16*   w0t  = (__bf16*)(ws + OFF_W0T);
    __bf16*   wst  = (__bf16*)(ws + OFF_WST);
    __bf16*   w1t  = (__bf16*)(ws + OFF_W1T);
    __bf16*   wct  = (__bf16*)(ws + OFF_WCT);
    int*      idx  = (int*)(ws + OFF_IDX);
    unsigned* grid = (unsigned*)(ws + OFF_GRID);
    float*    cnt  = (float*)(ws + OFF_CNT);

    float* out     = (float*)d_out;
    float* xyz_out = out;                       // B*N*3        = 393216
    float* c_out   = out + 393216;              // B*N*128      = 16777216
    float* tri_out = out + 393216 + 16777216;   // 3*B*128*64*64= 6291456

    // one-time weight transpose/convert (L2-resident afterwards)
    wprep_kernel<<<(5*256*128 + 255)/256, 256, 0, stream>>>(fc0_w, w0t, 5, 256, 128);
    wprep_kernel<<<(5*256*128 + 255)/256, 256, 0, stream>>>(sc_w,  wst, 5, 256, 128);
    wprep_kernel<<<(5*128*128 + 255)/256, 256, 0, stream>>>(fc1_w, w1t, 5, 128, 128);
    wprep_kernel<<<(128*128   + 255)/256, 256, 0, stream>>>(fcc_w, wct, 1, 128, 128);

    prep_points_kernel<<<BN/256, 256, 0, stream>>>(x, idx, xyz_out);
    stem_kernel<<<BN/8, 256, 0, stream>>>(x, stem_w, stem_b, hw);

    // block 0: input is stem output (width 256)
    block_kernel<<<BN/TM, 128, 0, stream>>>(hw, hn, w0t, fc0_b, w1t, fc1_b, wst, sc_b);

    for (int i = 1; i < NBLK; ++i) {
        hipMemsetAsync(grid, 0, (size_t)3 * B_ * R2 * 128 * 4, stream);  // 0 == -inf
        scatter_max_kernel<<<(3*BN*32)/256, 256, 0, stream>>>(hn, idx, grid);
        gather_concat_kernel<<<(BN*32)/256, 256, 0, stream>>>(grid, idx, hn, hw);
        block_kernel<<<BN/TM, 128, 0, stream>>>(
            hw, hn,
            w0t + (size_t)i * 256 * 128, fc0_b + i * 128,
            w1t + (size_t)i * 128 * 128, fc1_b + i * 128,
            wst + (size_t)i * 256 * 128, sc_b  + i * 128);
    }

    head_kernel<<<BN/TM, 128, 0, stream>>>(hn, wct, fcc_b, c_out);

    hipMemsetAsync(tri_out, 0, (size_t)6291456 * 4, stream);
    hipMemsetAsync(cnt, 0, (size_t)3 * B_ * R2 * 4, stream);
    tri_scatter_kernel<<<(3*BN*32)/256, 256, 0, stream>>>(c_out, idx, tri_out, cnt);
    tri_final_kernel<<<6291456/256, 256, 0, stream>>>(tri_out, cnt);
}
